// VQConvRnnAE_57775900065849
// MI455X (gfx1250) — matrix operand order repaired and checked by reading, not compiled, not run on hardware
//
#include <hip/hip_runtime.h>
#include <hip/hip_bf16.h>

// ---------------------------------------------------------------------------
// MI455X (gfx1250) VQ-conv-RNN autoencoder, wave32 WMMA f16->f32.
//
// Layout conventions chosen so every WMMA fragment is contiguous in memory:
//   * activations:  h1[b][p][c]  (position-major, c contiguous)
//   * A operands :  row-major [M][K]  (raw weight layouts)
//   * B operands :  TRANSPOSED [N][Kpad] (column's K-run contiguous)
// Per-position GEMMs run in (M=channels, N=positions) orientation so D-tile
// stores are contiguous 8x f16 runs per lane.
// ---------------------------------------------------------------------------

typedef __attribute__((ext_vector_type(16))) _Float16 v16h;
typedef __attribute__((ext_vector_type(8)))  float    v8f;
typedef _Float16 h16;

#define B_   256
#define T_   19

__device__ __forceinline__ v8f zero8() {
  v8f z = {0.f,0.f,0.f,0.f,0.f,0.f,0.f,0.f};
  return z;
}

__device__ __forceinline__ v8f wmma_f16(v16h a, v16h b, v8f c) {
  return __builtin_amdgcn_wmma_f32_16x16x32_f16(
      false, a, false, b, (short)0, c, false, false);
}

// A fragment from row-major [M][K] storage (LDS or global), rows contiguous.
// ISA 7.12.2: lanes 0-15 row M=lane&15, halves 0-7 -> K=kb..kb+7,
// halves 8-15 -> K=kb+16..kb+23; lanes 16-31 same rows, kb += 8.
__device__ __forceinline__ v16h frag_a_rm(const h16* s, long long row0, int k0,
                                          long long rs) {
  int lane = threadIdx.x & 31;
  long long r = row0 + (lane & 15);
  int kb = k0 + ((lane >> 4) << 3);
  const h16* base = s + r * rs + kb;
  v16h a;
#pragma unroll
  for (int i = 0; i < 8; ++i)  a[i] = base[i];        // contiguous -> b128
#pragma unroll
  for (int i = 8; i < 16; ++i) a[i] = base[8 + i];    // contiguous -> b128
  return a;
}

// B fragment from TRANSPOSED storage [N][ld]: column n's K-run contiguous.
// lane holds col N=col0+(lane&15); halves 0-7 K=kb..kb+7, 8-15 K=kb+16..kb+23; lanes 16-31 kb += 8.
__device__ __forceinline__ v16h frag_bt(const h16* s, long long col0, int k0,
                                        long long ld) {
  int lane = threadIdx.x & 31;
  long long c = col0 + (lane & 15);
  int kb = k0 + ((lane >> 4) << 3);
  const h16* base = s + c * ld + kb;
  v16h b;
#pragma unroll
  for (int i = 0; i < 16; ++i) b[i] = base[i < 8 ? i : 8 + i];  // 2x b128
  return b;
}

// ---------------------------------------------------------------------------
// Weight conversion.
// ---------------------------------------------------------------------------
__global__ void k_convert(const float* enc_w0, const float* enc_r0,
                          const float* enc_w1, const float* enc_lin_w,
                          const float* emb, const float* dec_lin_w,
                          const float* dec_r0, const float* dec_w1,
                          const float* dec_w0, const float* dec_r0_b,
                          const float* dec_w1_b,
                          h16* W0e, h16* R0e, h16* W1eT, h16* LINeT, h16* EmbT,
                          h16* LINdT, h16* R0d, h16* W1dT, h16* W0dT,
                          float* biasSum, float* embsq) {
  int i0 = blockIdx.x * blockDim.x + threadIdx.x;
  int st = gridDim.x * blockDim.x;
  // A-side conv weights [64 o][32 k], k = c*9+i*3+j (natural order, pad 27->32)
  for (int i = i0; i < 64 * 32; i += st) {
    int o = i >> 5, k = i & 31;
    W0e[i] = (h16)(k < 27 ? enc_w0[o * 27 + k] : 0.f);
  }
  // A-side enc_r0 [o][c]: raw cast
  for (int i = i0; i < 64 * 64; i += st) R0e[i] = (h16)enc_r0[i];
  // B^T contract weights [256 o][4096 k'], k' = p*64+c (position-major K)
  for (int i = i0; i < 4096 * 256; i += st) {
    int o = i >> 12, r = i & 4095, p = r >> 6, c = r & 63;
    W1eT[i] = (h16)enc_w1[o * 4096 + c * 64 + p];
  }
  // B^T enc_lin [n][k] == raw enc_lin_w
  for (int i = i0; i < 256 * 256; i += st) LINeT[i] = (h16)enc_lin_w[i];
  // B^T codebook [j][c] == raw emb
  for (int i = i0; i < 512 * 64; i += st) EmbT[i] = (h16)emb[i];
  // B^T dec_lin [n][k] == raw dec_lin_w
  for (int i = i0; i < 128 * 128; i += st) LINdT[i] = (h16)dec_lin_w[i];
  // A-side dec_r0 [o][c]: raw cast
  for (int i = i0; i < 64 * 64; i += st) R0d[i] = (h16)dec_r0[i];
  // B^T dec_w1 [n'=p*64+o][128 c]
  for (int i = i0; i < 4096 * 128; i += st) {
    int n = i >> 7, c = i & 127, p = n >> 6, o = n & 63;
    W1dT[i] = (h16)dec_w1[c * 4096 + o * 64 + p];
  }
  // B^T dec_w0 [32 n][64 c], n = o*9+i*3+j (pad 27->32)
  for (int i = i0; i < 32 * 64; i += st) {
    int n = i >> 6, c = i & 63;
    W0dT[i] = (h16)(n < 27 ? dec_w0[c * 27 + n] : 0.f);
  }
  for (int i = i0; i < 64; i += st) biasSum[i] = dec_r0_b[i] + dec_w1_b[i];
  for (int i = i0; i < 512; i += st) {
    float s = 0.f;
    for (int c = 0; c < 64; ++c) { float e = emb[i * 64 + c]; s += e * e; }
    embsq[i] = s;
  }
}

__global__ void k_init(float* out, float* counts) {
  int t = blockIdx.x * blockDim.x + threadIdx.x;
  if (t < 2)   out[t] = 0.f;
  if (t < 512) counts[t] = 0.f;
}

// ---------------------------------------------------------------------------
// Encoder h1 step, orientation M=o(64) x N=p(64), one workgroup per batch b.
//   h1_new[b][p][o] = relu(conv0(x_t)[o,p] + sum_c r0[o][c]*h1_old[b][p][c]
//                          + bw0[o] (+ br0[o]))
// ---------------------------------------------------------------------------
__global__ void k_enc_h1(const float* __restrict__ x, int t,
                         const h16* __restrict__ h1_old,
                         const h16* __restrict__ W0e, const h16* __restrict__ R0e,
                         const float* __restrict__ bw0,
                         const float* __restrict__ br0, int first,
                         h16* __restrict__ h1_new, float* __restrict__ h1_f32) {
  __shared__ h16 sBt[64 * 32];  // im2col^T: [p][k], k contiguous
  int tid = threadIdx.x, lane = tid & 31, wv = tid >> 5;
  long long b = blockIdx.x;
  const float* xb = x + (size_t)(b * T_ + t) * 1728;
  if (t < T_ - 1) __builtin_prefetch(xb + 1728, 0, 1);

  v8f acc[4];
#pragma unroll
  for (int nt = 0; nt < 4; ++nt) acc[nt] = zero8();

  // stage im2col B^T tile: sBt[p][k] = x[b,t,c, oh*3+i, ow*3+j]
  for (int idx = tid; idx < 2048; idx += 128) {
    int p = idx >> 5, k = idx & 31;
    h16 v = (h16)0.f;
    if (k < 27) {
      int c = k / 9, rem = k % 9, ii = rem / 3, jj = rem % 3;
      int oh = p >> 3, ow = p & 7;
      v = (h16)xb[c * 576 + (oh * 3 + ii) * 24 + (ow * 3 + jj)];
    }
    sBt[idx] = v;
  }
  __syncthreads();
  {
    v16h a = frag_a_rm(W0e, wv * 16, 0, 32);   // raw conv weights, global/L0
#pragma unroll
    for (int nt = 0; nt < 4; ++nt)
      acc[nt] = wmma_f16(a, frag_bt(sBt, nt * 16, 0, 32), acc[nt]);
  }
  // recurrent: B = h1_old[b] position-major, direct global (contiguous frags)
  if (!first) {
    const h16* Bp = h1_old + b * 4096;
#pragma unroll
    for (int k0 = 0; k0 < 64; k0 += 32) {
      v16h a = frag_a_rm(R0e, wv * 16, k0, 64);
#pragma unroll
      for (int nt = 0; nt < 4; ++nt)
        acc[nt] = wmma_f16(a, frag_bt(Bp, nt * 16, k0, 64), acc[nt]);
    }
  }
  // epilogue: lane holds fixed p, VGPRs run over consecutive o -> b128 stores
  int col = lane & 15, rb = (lane >> 4) << 3;
  int o0 = wv * 16 + rb;
#pragma unroll
  for (int nt = 0; nt < 4; ++nt) {
    int p = nt * 16 + col;
    long long base = b * 4096 + (long long)p * 64 + o0;
#pragma unroll
    for (int i = 0; i < 8; ++i) {
      float v = acc[nt][i] + bw0[o0 + i] + (first ? 0.f : br0[o0 + i]);
      v = fmaxf(v, 0.f);
      h1_new[base + i] = (h16)v;
      if (h1_f32) h1_f32[base + i] = v;
    }
  }
}

// ---------------------------------------------------------------------------
// Generic two-phase WMMA GEMM for batch-major GEMMs (A row-major [M][K],
// B transposed [N][Kpad]); 64x64 tile per workgroup, 4 wave32 waves.
// ---------------------------------------------------------------------------
struct GemmP {
  long long M, N;
  const h16* A0; long long rs0; const h16* B0; long long ldb0; int K0;
  const float* midbias;  // after phase0: acc = relu(acc + midbias[n])
  const h16* A1; long long rs1; const h16* B1; long long ldb1; int K1;
  const float* bias;
  int relu;
  h16* out16; float* out32; long long ldo;
};

__global__ void k_gemm2(GemmP p) {
  __shared__ h16 sA[64 * 32];   // [m][k]
  __shared__ h16 sBt[64 * 32];  // [n][k]
  int tid = threadIdx.x, lane = tid & 31, wv = tid >> 5;
  long long bm0 = (long long)blockIdx.x * 64;
  long long bn0 = (long long)blockIdx.y * 64;
  v8f acc[4];
#pragma unroll
  for (int nt = 0; nt < 4; ++nt) acc[nt] = zero8();

  for (int ph = 0; ph < 2; ++ph) {
    const h16* A = ph ? p.A1 : p.A0;
    if (!A) break;
    const h16* BT = ph ? p.B1 : p.B0;
    int K = ph ? p.K1 : p.K0;
    long long rs = ph ? p.rs1 : p.rs0;
    long long ldb = ph ? p.ldb1 : p.ldb0;
    for (int k0 = 0; k0 < K; k0 += 32) {
      // stage A tile: 64 rows x 32 halfs = 256 uint4
      for (int u = tid; u < 256; u += 128) {
        int r = u >> 2, q = u & 3;
        long long m = bm0 + r;
        uint4 val = make_uint4(0u, 0u, 0u, 0u);
        if (m < p.M)
          val = *(const uint4*)(A + m * rs + k0 + q * 8);
        ((uint4*)sA)[u] = val;
      }
      // stage B^T tile
      for (int u = tid; u < 256; u += 128) {
        int n = u >> 2, q = u & 3;
        long long nn = bn0 + n;
        uint4 val = make_uint4(0u, 0u, 0u, 0u);
        if (nn < p.N)
          val = *(const uint4*)(BT + nn * ldb + k0 + q * 8);
        ((uint4*)sBt)[u] = val;
      }
      __syncthreads();
      v16h a = frag_a_rm(sA, wv * 16, 0, 32);
#pragma unroll
      for (int nt = 0; nt < 4; ++nt)
        acc[nt] = wmma_f16(a, frag_bt(sBt, nt * 16, 0, 32), acc[nt]);
      __syncthreads();
    }
    if (ph == 0 && p.midbias) {
      int col = lane & 15;
#pragma unroll
      for (int nt = 0; nt < 4; ++nt) {
        long long n = bn0 + nt * 16 + col;
        float mb = (n < p.N) ? p.midbias[n] : 0.f;
#pragma unroll
        for (int i = 0; i < 8; ++i) acc[nt][i] = fmaxf(acc[nt][i] + mb, 0.f);
      }
    }
  }

  int col = lane & 15, rb = (lane >> 4) << 3;
  for (int nt = 0; nt < 4; ++nt) {
    long long n = bn0 + nt * 16 + col;
    if (n >= p.N) continue;
    float bs = p.bias ? p.bias[n] : 0.f;
    for (int i = 0; i < 8; ++i) {
      long long m = bm0 + wv * 16 + rb + i;
      if (m >= p.M) continue;
      long long off = m * p.ldo + n;
      float v = acc[nt][i] + bs;
      if (p.relu) v = fmaxf(v, 0.f);
      if (p.out16) p.out16[off] = (h16)v;
      if (p.out32) p.out32[off] = v;
    }
  }
}

// ---------------------------------------------------------------------------
// Decoder recurrent step, M=o(64) x N=p(64), per-b workgroup, LDS-free:
//   h1_new[b][p][o] = relu(h1l[b][p][o] + sum_c r0[o][c]*h1_old[b][p][c]
//                          + (dec_r0_b+dec_w1_b)[o])
// ---------------------------------------------------------------------------
__global__ void k_dec_h1h(const h16* __restrict__ h1_old,
                          const h16* __restrict__ R0d,
                          const float* __restrict__ h1l,
                          const float* __restrict__ bsum,
                          h16* __restrict__ h1_new) {
  int tid = threadIdx.x, lane = tid & 31, wv = tid >> 5;
  long long b = blockIdx.x;
  const h16* Bp = h1_old + b * 4096;
  v8f acc[4];
#pragma unroll
  for (int nt = 0; nt < 4; ++nt) acc[nt] = zero8();
#pragma unroll
  for (int k0 = 0; k0 < 64; k0 += 32) {
    v16h a = frag_a_rm(R0d, wv * 16, k0, 64);
#pragma unroll
    for (int nt = 0; nt < 4; ++nt)
      acc[nt] = wmma_f16(a, frag_bt(Bp, nt * 16, k0, 64), acc[nt]);
  }
  int col = lane & 15, rb = (lane >> 4) << 3;
  int o0 = wv * 16 + rb;
#pragma unroll
  for (int nt = 0; nt < 4; ++nt) {
    int p = nt * 16 + col;
    long long base = b * 4096 + (long long)p * 64 + o0;
#pragma unroll
    for (int i = 0; i < 8; ++i) {
      float v = acc[nt][i] + h1l[base + i] + bsum[o0 + i];
      h1_new[base + i] = (h16)fmaxf(v, 0.f);
    }
  }
}

// ---------------------------------------------------------------------------
// VQ argmin: WMMA distances (key = ||e_j||^2 - 2 flat.e_j), one block =
// 16 rows x 512 codes, direct-global contiguous fragments.
// ---------------------------------------------------------------------------
__global__ void k_vq_argmin(const h16* __restrict__ h1,
                            const h16* __restrict__ EmbT,
                            const float* __restrict__ embsq,
                            int* __restrict__ idxo, float* __restrict__ counts) {
  int tid = threadIdx.x, lane = tid & 31, wv = tid >> 5;
  long long m0 = (long long)blockIdx.x * 16;
  v8f acc[8];
#pragma unroll
  for (int nt = 0; nt < 8; ++nt) acc[nt] = zero8();
#pragma unroll
  for (int k0 = 0; k0 < 64; k0 += 32) {
    v16h a = frag_a_rm(h1, m0, k0, 64);
#pragma unroll
    for (int nt = 0; nt < 8; ++nt)
      acc[nt] = wmma_f16(a, frag_bt(EmbT, wv * 128 + nt * 16, k0, 64), acc[nt]);
  }
  __shared__ float sv[16][4];
  __shared__ int   si[16][4];
  int col = lane & 15;
#pragma unroll
  for (int i = 0; i < 8; ++i) {
    float best = 3.4e38f;
    int bidx = 0;
#pragma unroll
    for (int nt = 0; nt < 8; ++nt) {
      int n = wv * 128 + nt * 16 + col;
      float d = embsq[n] - 2.f * acc[nt][i];
      if (d < best) { best = d; bidx = n; }
    }
#pragma unroll
    for (int off = 8; off >= 1; off >>= 1) {
      float ov = __shfl_xor(best, off, 16);
      int   oi = __shfl_xor(bidx, off, 16);
      if (ov < best || (ov == best && oi < bidx)) { best = ov; bidx = oi; }
    }
    int row = ((lane >> 4) << 3) + i;
    if (col == 0) { sv[row][wv] = best; si[row][wv] = bidx; }
  }
  __syncthreads();
  if (tid < 16) {
    float best = sv[tid][0];
    int bidx = si[tid][0];
    for (int w = 1; w < 4; ++w)
      if (sv[tid][w] < best || (sv[tid][w] == best && si[tid][w] < bidx)) {
        best = sv[tid][w]; bidx = si[tid][w];
      }
    idxo[m0 + tid] = bidx;
    atomicAdd(&counts[bidx], 1.f);
  }
}

__global__ void k_dequant(const int* __restrict__ idx,
                          const float* __restrict__ emb,
                          h16* __restrict__ h1dec) {
  int m = blockIdx.x * blockDim.x + threadIdx.x;
  if (m >= B_ * 64) return;
  int j = idx[m];
  long long base = (long long)m * 64;
  for (int c = 0; c < 64; ++c) h1dec[base + c] = (h16)emb[j * 64 + c];
}

__global__ void k_vq_loss(const int* __restrict__ idx,
                          const float* __restrict__ emb,
                          const float* __restrict__ h1f,
                          float* __restrict__ out0) {
  __shared__ float red[256];
  float s = 0.f;
  for (int m = blockIdx.x * blockDim.x + threadIdx.x; m < B_ * 64;
       m += gridDim.x * blockDim.x) {
    int j = idx[m];
    long long base = (long long)m * 64;
    for (int c = 0; c < 64; ++c) {
      float dd = emb[j * 64 + c] - h1f[base + c];
      s += dd * dd;
    }
  }
  red[threadIdx.x] = s;
  __syncthreads();
  for (int o = blockDim.x >> 1; o > 0; o >>= 1) {
    if (threadIdx.x < (unsigned)o) red[threadIdx.x] += red[threadIdx.x + o];
    __syncthreads();
  }
  if (threadIdx.x == 0)
    atomicAdd(out0, red[0] * (0.25f / (16384.f * 64.f)));
}

__global__ void k_kl(const float* __restrict__ h2f, const float* __restrict__ eps,
                     float* __restrict__ out1, h16* __restrict__ h2s) {
  int b = blockIdx.x, n = threadIdx.x;  // 128 threads
  float mu = h2f[b * 256 + n];
  float lv = h2f[b * 256 + 128 + n];
  float term = 0.5f * (expf(lv) + mu * mu - 1.f - lv);
  h2s[b * 128 + n] = (h16)(mu + expf(0.5f * lv) * eps[b * 128 + n]);
  __shared__ float red[128];
  red[n] = term;
  __syncthreads();
  for (int o = 64; o > 0; o >>= 1) {
    if (n < o) red[n] += red[n + o];
    __syncthreads();
  }
  if (n == 0) atomicAdd(out1, red[0] / 256.f);
}

__global__ void k_perp(const float* __restrict__ counts, float* __restrict__ outp) {
  __shared__ float red[512];
  int t = threadIdx.x;
  float avg = counts[t] / 16384.f;
  red[t] = avg * logf(avg + 1e-10f);
  __syncthreads();
  for (int o = 256; o > 0; o >>= 1) {
    if (t < o) red[t] += red[t + o];
    __syncthreads();
  }
  if (t == 0) *outp = expf(-red[0]);
}

// ---------------------------------------------------------------------------
// Decoder output conv-transpose: M=p rows (contiguous A), N=27 pad 32, K=64.
//   y[b,o,3h+i,3w+j] = elu(sum_c h1[b][p][c]*W0dT[n][c] + b0[o])
// ---------------------------------------------------------------------------
__global__ void k_dec_out(const h16* __restrict__ h1, const h16* __restrict__ W0dT,
                          const float* __restrict__ b0, int t,
                          float* __restrict__ out) {
  int tid = threadIdx.x, lane = tid & 31, wv = tid >> 5;
  long long b = blockIdx.x;
  long long m0 = b * 64 + wv * 16;
  v8f acc[2];
  acc[0] = zero8();
  acc[1] = zero8();
#pragma unroll
  for (int k0 = 0; k0 < 64; k0 += 32) {
    v16h a = frag_a_rm(h1, m0, k0, 64);
#pragma unroll
    for (int nt = 0; nt < 2; ++nt)
      acc[nt] = wmma_f16(a, frag_bt(W0dT, nt * 16, k0, 64), acc[nt]);
  }
  int col = lane & 15, rb = (lane >> 4) << 3;
  for (int nt = 0; nt < 2; ++nt) {
    int n = nt * 16 + col;
    if (n >= 27) continue;
    int o = n / 9, rem = n % 9, ii = rem / 3, jj = rem % 3;
    float bias = b0[o];
    for (int i = 0; i < 8; ++i) {
      int p = wv * 16 + rb + i;
      int oh = p >> 3, ow = p & 7;
      float v = acc[nt][i] + bias;
      v = v > 0.f ? v : (expf(v) - 1.f);
      out[((size_t)(b * T_ + t) * 3 + o) * 576 + (oh * 3 + ii) * 24 +
          (ow * 3 + jj)] = v;
    }
  }
}

// ---------------------------------------------------------------------------
// Host orchestration
// ---------------------------------------------------------------------------
extern "C" void kernel_launch(void* const* d_in, const int* in_sizes, int n_in,
                              void* d_out, int out_size, void* d_ws,
                              size_t ws_size, hipStream_t stream) {
  (void)in_sizes; (void)n_in; (void)out_size; (void)ws_size;
  const float* x         = (const float*)d_in[0];
  const float* eps       = (const float*)d_in[1];
  const float* emb       = (const float*)d_in[2];
  const float* enc_w0    = (const float*)d_in[3];
  const float* enc_w0_b  = (const float*)d_in[4];
  const float* enc_w1    = (const float*)d_in[5];
  const float* enc_w1_b  = (const float*)d_in[6];
  const float* enc_r0    = (const float*)d_in[7];
  const float* enc_r0_b  = (const float*)d_in[8];
  const float* enc_lin_w = (const float*)d_in[9];
  const float* enc_lin_b = (const float*)d_in[10];
  const float* dec_w0    = (const float*)d_in[11];
  const float* dec_w0_b  = (const float*)d_in[12];
  const float* dec_w1    = (const float*)d_in[13];
  const float* dec_w1_b  = (const float*)d_in[14];
  const float* dec_r0    = (const float*)d_in[15];
  const float* dec_r0_b  = (const float*)d_in[16];
  const float* dec_lin_w = (const float*)d_in[17];
  const float* dec_lin_b = (const float*)d_in[18];
  float* out = (float*)d_out;

  char* w = (char*)d_ws;
  auto alloc = [&](size_t bytes) -> char* {
    char* p = w;
    w += (bytes + 255) & ~(size_t)255;
    return p;
  };
  h16* W0e   = (h16*)alloc(64 * 32 * 2);
  h16* R0e   = (h16*)alloc(64 * 64 * 2);
  h16* W1eT  = (h16*)alloc((size_t)4096 * 256 * 2);
  h16* LINeT = (h16*)alloc(256 * 256 * 2);
  h16* EmbT  = (h16*)alloc(512 * 64 * 2);
  h16* LINdT = (h16*)alloc(128 * 128 * 2);
  h16* R0d   = (h16*)alloc(64 * 64 * 2);
  h16* W1dT  = (h16*)alloc((size_t)4096 * 128 * 2);
  h16* W0dT  = (h16*)alloc(32 * 64 * 2);
  float* biasSum = (float*)alloc(64 * 4);
  float* embsq   = (float*)alloc(512 * 4);
  float* counts  = (float*)alloc(512 * 4);
  int*   idxbuf  = (int*)alloc((size_t)B_ * 64 * 4);
  h16* h1a   = (h16*)alloc((size_t)B_ * 4096 * 2);
  h16* h1b   = (h16*)alloc((size_t)B_ * 4096 * 2);
  float* h1f = (float*)alloc((size_t)B_ * 4096 * 4);
  h16* h2a   = (h16*)alloc((size_t)B_ * 256 * 2);
  h16* h2b   = (h16*)alloc((size_t)B_ * 256 * 2);
  float* h2f = (float*)alloc((size_t)B_ * 256 * 4);
  h16* h2d_a = (h16*)alloc((size_t)B_ * 128 * 2);
  h16* h2d_b = (h16*)alloc((size_t)B_ * 128 * 2);
  float* h1l = (float*)alloc((size_t)B_ * 4096 * 4);
  h16* h1d_a = (h16*)alloc((size_t)B_ * 4096 * 2);
  h16* h1d_b = (h16*)alloc((size_t)B_ * 4096 * 2);

  k_convert<<<2048, 256, 0, stream>>>(enc_w0, enc_r0, enc_w1, enc_lin_w, emb,
                                      dec_lin_w, dec_r0, dec_w1, dec_w0,
                                      dec_r0_b, dec_w1_b, W0e, R0e, W1eT, LINeT,
                                      EmbT, LINdT, R0d, W1dT, W0dT, biasSum,
                                      embsq);
  k_init<<<2, 256, 0, stream>>>(out, counts);

  // ---- encoder recurrence over 19 frames ----
  h16* h1cur = h1a;
  h16* h2cur = h2a;
  k_enc_h1<<<256, 128, 0, stream>>>(x, 0, nullptr, W0e, R0e, enc_w0_b,
                                    enc_r0_b, 1, h1cur, nullptr);
  {
    GemmP g{};
    g.M = 256; g.N = 256;
    g.A0 = h1cur; g.rs0 = 4096; g.B0 = W1eT; g.ldb0 = 4096; g.K0 = 4096;
    g.midbias = enc_w1_b;
    g.relu = 1; g.out16 = h2cur; g.out32 = h2f; g.ldo = 256;
    k_gemm2<<<dim3(4, 4), 128, 0, stream>>>(g);
  }
  for (int t = 1; t < T_; ++t) {
    h16* h1next = (h1cur == h1a) ? h1b : h1a;
    k_enc_h1<<<256, 128, 0, stream>>>(x, t, h1cur, W0e, R0e, enc_w0_b,
                                      enc_r0_b, 0, h1next,
                                      (t == T_ - 1) ? h1f : nullptr);
    h1cur = h1next;
    h16* h2next = (h2cur == h2a) ? h2b : h2a;
    GemmP g{};
    g.M = 256; g.N = 256;
    g.A0 = h1cur; g.rs0 = 4096; g.B0 = W1eT; g.ldb0 = 4096; g.K0 = 4096;
    g.midbias = enc_w1_b;
    g.A1 = h2cur; g.rs1 = 256; g.B1 = LINeT; g.ldb1 = 256; g.K1 = 256;
    g.bias = enc_lin_b;
    g.relu = 1; g.out16 = h2next; g.out32 = h2f; g.ldo = 256;
    k_gemm2<<<dim3(4, 4), 128, 0, stream>>>(g);
    h2cur = h2next;
  }

  // ---- VQ + KL ----
  k_vq_argmin<<<1024, 128, 0, stream>>>(h1cur, EmbT, embsq, idxbuf, counts);
  k_vq_loss<<<64, 256, 0, stream>>>(idxbuf, emb, h1f, out + 0);
  k_kl<<<256, 128, 0, stream>>>(h2f, eps, out + 1, h2d_a);
  k_dequant<<<128, 128, 0, stream>>>(idxbuf, emb, h1d_a);
  k_perp<<<1, 512, 0, stream>>>(counts, out + 2 + (size_t)B_ * T_ * 3 * 576);

  // ---- decoder: 1 warm-up update + 19 (update, emit) steps ----
  h16* dh2 = h2d_a;
  h16* dh1 = h1d_a;
  auto dec_update = [&]() {
    h16* h2n = (dh2 == h2d_a) ? h2d_b : h2d_a;
    GemmP gl{};
    gl.M = 256; gl.N = 128;
    gl.A0 = dh2; gl.rs0 = 128; gl.B0 = LINdT; gl.ldb0 = 128; gl.K0 = 128;
    gl.bias = dec_lin_b; gl.relu = 1; gl.out16 = h2n; gl.ldo = 128;
    k_gemm2<<<dim3(4, 2), 128, 0, stream>>>(gl);

    GemmP gb{};
    gb.M = 256; gb.N = 4096;
    gb.A0 = h2n; gb.rs0 = 128; gb.B0 = W1dT; gb.ldb0 = 128; gb.K0 = 128;
    gb.out32 = h1l; gb.ldo = 4096;
    k_gemm2<<<dim3(4, 64), 128, 0, stream>>>(gb);

    h16* h1n = (dh1 == h1d_a) ? h1d_b : h1d_a;
    k_dec_h1h<<<256, 128, 0, stream>>>(dh1, R0d, h1l, biasSum, h1n);
    dh2 = h2n;
    dh1 = h1n;
  };
  dec_update();
  for (int s = 0; s < T_; ++s) {
    dec_update();
    k_dec_out<<<256, 128, 0, stream>>>(dh1, W0dT, dec_w0_b, s, out + 2);
  }
}
